// SeqVLADModule_274877907331
// MI455X (gfx1250) — matrix operand order repaired and checked by reading, not compiled, not run on hardware
//
#include <hip/hip_runtime.h>
#include <hip/hip_bf16.h>
#include <math.h>

typedef __bf16 bf16_t;
typedef __attribute__((ext_vector_type(16))) __bf16 v16bf;
typedef __attribute__((ext_vector_type(4)))  __bf16 v4bf;
typedef __attribute__((ext_vector_type(8)))  float  v8f;

#define T_STEPS 25
#define B_IMG   8
#define NTOT    200      // B*T
#define C_IN    1024
#define D_DIM   512
#define K_DIM   64
#define HW      196
#define WD      14
#define NPIX    (B_IMG*HW)   // 1568

// ---- fp32 -> bf16 (round to nearest even) -------------------------------
__device__ __forceinline__ bf16_t f2bf(float f) {
  unsigned u = __float_as_uint(f);
  u += 0x7FFFu + ((u >> 16) & 1u);
  unsigned short h = (unsigned short)(u >> 16);
  return __builtin_bit_cast(bf16_t, h);
}
__device__ __forceinline__ v4bf f2bf4(float a, float b, float c, float d) {
  v4bf p; p[0] = f2bf(a); p[1] = f2bf(b); p[2] = f2bf(c); p[3] = f2bf(d);
  return p;
}

// ---- fragment-order LDS addressing (contiguous k order, k = 16*(lane>>4) + idx, same for A and B) ----------
// A 16x32 bf16 tile: lane = (m&15) + (k>>4)*16 ; idx = k&15
__device__ __forceinline__ int a_base(int m, int k) {
  return (m >> 4) * 512 + (((m & 15) + ((k >> 4) << 4)) << 4) +
         (k & 15);
}
// B 32x16 bf16 tile: lane = (j&15) + (k>>4)*16 ; idx = k&15
__device__ __forceinline__ int b_base(int k, int j) {
  return (j >> 4) * 512 + (((j & 15) + ((k >> 4) << 4)) << 4) + (k & 15);
}
// one lane's whole fragment is 16 contiguous bf16 (32B) -> 2x ds_load_b128
__device__ __forceinline__ v16bf frag16(const bf16_t* p, int tile, int lane) {
  return *(const v16bf*)(p + ((tile * 32 + lane) << 4));
}
__device__ __forceinline__ v8f wmma_bf16(v16bf a, v16bf b, v8f c) {
  return __builtin_amdgcn_wmma_f32_16x16x32_bf16(false, a, false, b,
                                                 (short)0, c, false, false);
}
// C/D 16x16 f32: VGPR r -> row r + (lane>=16 ? 8 : 0), col = lane&15.

__device__ __forceinline__ float sigmoidf_(float x) {
  return 1.0f / (1.0f + __expf(-x));
}

// =========================================================================
// Kernel 1: feat = redu_w(512x1024) @ x[n](1024x196) + redu_b, bf16 out
// block tile 64(m) x 128(s); wave tile 32x32 (4 WMMA / K-step)
// grid (8 mtiles, 2 stiles, 200)
// =========================================================================
__global__ __launch_bounds__(256) void k_feat(const float* __restrict__ x,
                                              const float* __restrict__ redu_w,
                                              const float* __restrict__ redu_b,
                                              bf16_t* __restrict__ feat) {
  __shared__ __align__(32) bf16_t As[64 * 32];
  __shared__ __align__(32) bf16_t Bs[128 * 32];
  const int m0 = blockIdx.x * 64;
  const int s0 = blockIdx.y * 128;
  const int n  = blockIdx.z;
  const int tid = threadIdx.x, lane = tid & 31, wave = tid >> 5;
  const int wmb = (wave >> 2) * 32;  // wave m base (0/32)
  const int wnb = (wave & 3) * 32;   // wave n base (0..96)
  const float* xn = x + (size_t)n * C_IN * HW;
  v8f acc00 = {}, acc01 = {}, acc10 = {}, acc11 = {};
  for (int k0 = 0; k0 < C_IN; k0 += 32) {
    // stage A (64x32): float4 along k -> one v4bf (b64) LDS store
#pragma unroll
    for (int g = tid; g < 512; g += 256) {
      int m = g >> 3, k = (g & 7) << 2;
      float4 v = *(const float4*)(redu_w + (size_t)(m0 + m) * C_IN + k0 + k);
      *(v4bf*)(As + a_base(m, k)) = f2bf4(v.x, v.y, v.z, v.w);
    }
    // stage B (32x128): float4 along s
#pragma unroll
    for (int g = tid; g < 1024; g += 256) {
      int k = g >> 5, j = (g & 31) << 2;
      int s = s0 + j;
      float4 v = {0.f, 0.f, 0.f, 0.f};
      if (s < HW) v = *(const float4*)(xn + (size_t)(k0 + k) * HW + s);
      Bs[b_base(k, j)]     = f2bf(v.x);
      Bs[b_base(k, j + 1)] = f2bf(v.y);
      Bs[b_base(k, j + 2)] = f2bf(v.z);
      Bs[b_base(k, j + 3)] = f2bf(v.w);
    }
    __syncthreads();
    v16bf a0 = frag16(As, (wmb >> 4),     lane);
    v16bf a1 = frag16(As, (wmb >> 4) + 1, lane);
    v16bf b0 = frag16(Bs, (wnb >> 4),     lane);
    v16bf b1 = frag16(Bs, (wnb >> 4) + 1, lane);
    acc00 = wmma_bf16(a0, b0, acc00);
    acc01 = wmma_bf16(a0, b1, acc01);
    acc10 = wmma_bf16(a1, b0, acc10);
    acc11 = wmma_bf16(a1, b1, acc11);
    __syncthreads();
  }
  const int rm = (lane >> 4) << 3, cn = lane & 15;
#pragma unroll
  for (int mi = 0; mi < 2; ++mi) {
#pragma unroll
    for (int ni = 0; ni < 2; ++ni) {
      v8f a = (mi == 0) ? (ni == 0 ? acc00 : acc01)
                        : (ni == 0 ? acc10 : acc11);
      int mb = m0 + wmb + mi * 16 + rm;
      int s  = s0 + wnb + ni * 16 + cn;
      if (s < HW) {
#pragma unroll
        for (int r = 0; r < 8; ++r) {
          int m = mb + r;
          feat[(size_t)n * D_DIM * HW + (size_t)m * HW + s] =
              f2bf(a[r] + redu_b[m]);
        }
      }
    }
  }
}

// =========================================================================
// Kernel 2: wx = share_w(64x512) @ feat[n](512x196) + share_b -> [t][b][k][s]
// grid (2 stiles, 200)
// =========================================================================
__global__ __launch_bounds__(256) void k_wx(const bf16_t* __restrict__ feat,
                                            const float* __restrict__ share_w,
                                            const float* __restrict__ share_b,
                                            float* __restrict__ wx) {
  __shared__ __align__(32) bf16_t As[64 * 32];
  __shared__ __align__(32) bf16_t Bs[128 * 32];
  const int s0 = blockIdx.x * 128;
  const int n  = blockIdx.y;
  const int tid = threadIdx.x, lane = tid & 31, wave = tid >> 5;
  const int wmb = (wave >> 2) * 32, wnb = (wave & 3) * 32;
  const bf16_t* fn = feat + (size_t)n * D_DIM * HW;
  v8f acc00 = {}, acc01 = {}, acc10 = {}, acc11 = {};
  for (int k0 = 0; k0 < D_DIM; k0 += 32) {
#pragma unroll
    for (int g = tid; g < 512; g += 256) {
      int m = g >> 3, k = (g & 7) << 2;
      float4 v = *(const float4*)(share_w + (size_t)m * D_DIM + k0 + k);
      *(v4bf*)(As + a_base(m, k)) = f2bf4(v.x, v.y, v.z, v.w);
    }
#pragma unroll
    for (int g = tid; g < 1024; g += 256) {
      int k = g >> 5, j = (g & 31) << 2;
      int s = s0 + j;
      v4bf v = {};
      if (s < HW) v = *(const v4bf*)(fn + (size_t)(k0 + k) * HW + s);
      Bs[b_base(k, j)]     = v[0];
      Bs[b_base(k, j + 1)] = v[1];
      Bs[b_base(k, j + 2)] = v[2];
      Bs[b_base(k, j + 3)] = v[3];
    }
    __syncthreads();
    v16bf a0 = frag16(As, (wmb >> 4),     lane);
    v16bf a1 = frag16(As, (wmb >> 4) + 1, lane);
    v16bf b0 = frag16(Bs, (wnb >> 4),     lane);
    v16bf b1 = frag16(Bs, (wnb >> 4) + 1, lane);
    acc00 = wmma_bf16(a0, b0, acc00);
    acc01 = wmma_bf16(a0, b1, acc01);
    acc10 = wmma_bf16(a1, b0, acc10);
    acc11 = wmma_bf16(a1, b1, acc11);
    __syncthreads();
  }
  const int b = n / T_STEPS, t = n % T_STEPS;
  const int rm = (lane >> 4) << 3, cn = lane & 15;
#pragma unroll
  for (int mi = 0; mi < 2; ++mi) {
#pragma unroll
    for (int ni = 0; ni < 2; ++ni) {
      v8f a = (mi == 0) ? (ni == 0 ? acc00 : acc01)
                        : (ni == 0 ? acc10 : acc11);
      int mb = wmb + mi * 16 + rm;
      int s  = s0 + wnb + ni * 16 + cn;
      if (s < HW) {
#pragma unroll
        for (int r = 0; r < 8; ++r) {
          int m = mb + r;
          wx[(((size_t)t * B_IMG + b) * K_DIM + m) * HW + s] =
              a[r] + share_b[m];
        }
      }
    }
  }
}

// =========================================================================
// GRU part 1: z = sig(wx + Uz*h), r = sig(wx + Ur*h); store z, r*h.
// Implicit GEMM M=64, K=576, N=1568 px; shared im2col tile, 8 WMMA / K-step.
// grid 13 (pixel tiles of 128)
// =========================================================================
__global__ __launch_bounds__(256) void k_gru_zr(const float* __restrict__ hist,
                                                const float* __restrict__ Uz,
                                                const float* __restrict__ Ur,
                                                const float* __restrict__ wx,
                                                int t,
                                                float* __restrict__ zbuf,
                                                float* __restrict__ rhbuf) {
  __shared__ __align__(32) bf16_t Az[64 * 32];
  __shared__ __align__(32) bf16_t Ar[64 * 32];
  __shared__ __align__(32) bf16_t Bs[128 * 32];
  const int p0 = blockIdx.x * 128;
  const int tid = threadIdx.x, lane = tid & 31, wave = tid >> 5;
  const int wmb = (wave >> 2) * 32, wnb = (wave & 3) * 32;
  const int tprev = t - 1;
  v8f az00 = {}, az01 = {}, az10 = {}, az11 = {};
  v8f ar00 = {}, ar01 = {}, ar10 = {}, ar11 = {};
  for (int k0 = 0; k0 < 576; k0 += 32) {
#pragma unroll
    for (int g = tid; g < 512; g += 256) {
      int m = g >> 3, k = (g & 7) << 2;
      float4 vz = *(const float4*)(Uz + (size_t)m * 576 + k0 + k);
      float4 vr = *(const float4*)(Ur + (size_t)m * 576 + k0 + k);
      *(v4bf*)(Az + a_base(m, k)) = f2bf4(vz.x, vz.y, vz.z, vz.w);
      *(v4bf*)(Ar + a_base(m, k)) = f2bf4(vr.x, vr.y, vr.z, vr.w);
    }
    // im2col B (32x128), elementwise (halo + batch bounds)
#pragma unroll
    for (int g = tid; g < 4096; g += 256) {
      int k = g >> 7, j = g & 127;
      int kk = k0 + k;
      int ci = kk / 9, rem = kk - ci * 9;
      int dy = rem / 3, dx = rem - dy * 3;
      int p = p0 + j;
      float v = 0.0f;
      if (p < NPIX && tprev >= 0) {
        int b = p / HW, s = p - b * HW;
        int iy = s / WD + dy - 1, ix = s - (s / WD) * WD + dx - 1;
        if (iy >= 0 && iy < WD && ix >= 0 && ix < WD)
          v = hist[(((size_t)b * T_STEPS + tprev) * K_DIM + ci) * HW +
                   iy * WD + ix];
      }
      Bs[b_base(k, j)] = f2bf(v);
    }
    __syncthreads();
    v16bf b0 = frag16(Bs, (wnb >> 4),     lane);
    v16bf b1 = frag16(Bs, (wnb >> 4) + 1, lane);
    v16bf z0 = frag16(Az, (wmb >> 4),     lane);
    v16bf z1 = frag16(Az, (wmb >> 4) + 1, lane);
    az00 = wmma_bf16(z0, b0, az00);
    az01 = wmma_bf16(z0, b1, az01);
    az10 = wmma_bf16(z1, b0, az10);
    az11 = wmma_bf16(z1, b1, az11);
    v16bf r0 = frag16(Ar, (wmb >> 4),     lane);
    v16bf r1 = frag16(Ar, (wmb >> 4) + 1, lane);
    ar00 = wmma_bf16(r0, b0, ar00);
    ar01 = wmma_bf16(r0, b1, ar01);
    ar10 = wmma_bf16(r1, b0, ar10);
    ar11 = wmma_bf16(r1, b1, ar11);
    __syncthreads();
  }
  const int rm = (lane >> 4) << 3, cn = lane & 15;
#pragma unroll
  for (int mi = 0; mi < 2; ++mi) {
#pragma unroll
    for (int ni = 0; ni < 2; ++ni) {
      v8f vz = (mi == 0) ? (ni == 0 ? az00 : az01) : (ni == 0 ? az10 : az11);
      v8f vr = (mi == 0) ? (ni == 0 ? ar00 : ar01) : (ni == 0 ? ar10 : ar11);
      int mb = wmb + mi * 16 + rm;
      int p  = p0 + wnb + ni * 16 + cn;
      if (p < NPIX) {
        int b = p / HW, s = p - (p / HW) * HW;
#pragma unroll
        for (int r = 0; r < 8; ++r) {
          int m = mb + r;
          size_t cidx = ((size_t)b * K_DIM + m) * HW + s;
          float wv = wx[(((size_t)t * B_IMG + b) * K_DIM + m) * HW + s];
          float z  = sigmoidf_(wv + vz[r]);
          float rr = sigmoidf_(wv + vr[r]);
          float hv = (tprev >= 0)
              ? hist[(((size_t)b * T_STEPS + tprev) * K_DIM + m) * HW + s]
              : 0.0f;
          zbuf[cidx]  = z;
          rhbuf[cidx] = rr * hv;
        }
      }
    }
  }
}

// =========================================================================
// GRU part 2: hh = tanh(wx + Uh*(r*h)); h_new = (1-z)*hh + z*h -> hist[b,t]
// =========================================================================
__global__ __launch_bounds__(256) void k_gru_h(float* __restrict__ hist,
                                               const float* __restrict__ Uh,
                                               const float* __restrict__ wx,
                                               int t,
                                               const float* __restrict__ zbuf,
                                               const float* __restrict__ rhbuf) {
  __shared__ __align__(32) bf16_t As[64 * 32];
  __shared__ __align__(32) bf16_t Bs[128 * 32];
  const int p0 = blockIdx.x * 128;
  const int tid = threadIdx.x, lane = tid & 31, wave = tid >> 5;
  const int wmb = (wave >> 2) * 32, wnb = (wave & 3) * 32;
  const int tprev = t - 1;
  v8f acc00 = {}, acc01 = {}, acc10 = {}, acc11 = {};
  for (int k0 = 0; k0 < 576; k0 += 32) {
#pragma unroll
    for (int g = tid; g < 512; g += 256) {
      int m = g >> 3, k = (g & 7) << 2;
      float4 v = *(const float4*)(Uh + (size_t)m * 576 + k0 + k);
      *(v4bf*)(As + a_base(m, k)) = f2bf4(v.x, v.y, v.z, v.w);
    }
#pragma unroll
    for (int g = tid; g < 4096; g += 256) {
      int k = g >> 7, j = g & 127;
      int kk = k0 + k;
      int ci = kk / 9, rem = kk - ci * 9;
      int dy = rem / 3, dx = rem - dy * 3;
      int p = p0 + j;
      float v = 0.0f;
      if (p < NPIX) {
        int b = p / HW, s = p - b * HW;
        int iy = s / WD + dy - 1, ix = s - (s / WD) * WD + dx - 1;
        if (iy >= 0 && iy < WD && ix >= 0 && ix < WD)
          v = rhbuf[((size_t)b * K_DIM + ci) * HW + iy * WD + ix];
      }
      Bs[b_base(k, j)] = f2bf(v);
    }
    __syncthreads();
    v16bf a0 = frag16(As, (wmb >> 4),     lane);
    v16bf a1 = frag16(As, (wmb >> 4) + 1, lane);
    v16bf b0 = frag16(Bs, (wnb >> 4),     lane);
    v16bf b1 = frag16(Bs, (wnb >> 4) + 1, lane);
    acc00 = wmma_bf16(a0, b0, acc00);
    acc01 = wmma_bf16(a0, b1, acc01);
    acc10 = wmma_bf16(a1, b0, acc10);
    acc11 = wmma_bf16(a1, b1, acc11);
    __syncthreads();
  }
  const int rm = (lane >> 4) << 3, cn = lane & 15;
#pragma unroll
  for (int mi = 0; mi < 2; ++mi) {
#pragma unroll
    for (int ni = 0; ni < 2; ++ni) {
      v8f a = (mi == 0) ? (ni == 0 ? acc00 : acc01)
                        : (ni == 0 ? acc10 : acc11);
      int mb = wmb + mi * 16 + rm;
      int p  = p0 + wnb + ni * 16 + cn;
      if (p < NPIX) {
        int b = p / HW, s = p - (p / HW) * HW;
#pragma unroll
        for (int r = 0; r < 8; ++r) {
          int m = mb + r;
          float wv = wx[(((size_t)t * B_IMG + b) * K_DIM + m) * HW + s];
          float hh = tanhf(wv + a[r]);
          float z  = zbuf[((size_t)b * K_DIM + m) * HW + s];
          float hv = (tprev >= 0)
              ? hist[(((size_t)b * T_STEPS + tprev) * K_DIM + m) * HW + s]
              : 0.0f;
          hist[(((size_t)b * T_STEPS + t) * K_DIM + m) * HW + s] =
              (1.0f - z) * hh + z * hv;
        }
      }
    }
  }
}

// =========================================================================
// Softmax over K=64 per (n, s); in-place on assigns [n][k][s].
// =========================================================================
__global__ __launch_bounds__(256) void k_softmax(float* __restrict__ assigns) {
  int gid = blockIdx.x * 256 + threadIdx.x;
  if (gid >= NTOT * HW) return;
  int n = gid / HW, s = gid - (gid / HW) * HW;
  float* base = assigns + (size_t)n * K_DIM * HW + s;
  float v[K_DIM];
  float mx = -3.0e38f;
#pragma unroll
  for (int k = 0; k < K_DIM; ++k) { v[k] = base[(size_t)k * HW]; mx = fmaxf(mx, v[k]); }
  float sum = 0.0f;
#pragma unroll
  for (int k = 0; k < K_DIM; ++k) { v[k] = __expf(v[k] - mx); sum += v[k]; }
  float inv = 1.0f / sum;
#pragma unroll
  for (int k = 0; k < K_DIM; ++k) base[(size_t)k * HW] = v[k] * inv;
}

// asum[n][k] = sum_s assigns[n][k][s]
__global__ __launch_bounds__(256) void k_asum(const float* __restrict__ assigns,
                                              float* __restrict__ asum) {
  int gid = blockIdx.x * 256 + threadIdx.x;
  if (gid >= NTOT * K_DIM) return;
  const float* row = assigns + (size_t)gid * HW;
  float ssum = 0.0f;
  for (int s = 0; s < HW; ++s) ssum += row[s];
  asum[gid] = ssum;
}

// =========================================================================
// VLAD: vlad[b,k,d] = sum_t (assigns[n] @ feat[n]^T)[k,d] - (sum_t asum)*centers
// A = assigns[n] (64x196), B^T from feat[n] (d-major), accumulate t & s chunks.
// grid (4 dtiles of 128, 8 batch)
// =========================================================================
__global__ __launch_bounds__(256) void k_vlad(const float* __restrict__ assigns,
                                              const bf16_t* __restrict__ feat,
                                              const float* __restrict__ asum,
                                              const float* __restrict__ centers,
                                              float* __restrict__ vlad) {
  __shared__ __align__(32) bf16_t As[64 * 32];
  __shared__ __align__(32) bf16_t Bs[128 * 32];
  const int d0 = blockIdx.x * 128;
  const int b  = blockIdx.y;
  const int tid = threadIdx.x, lane = tid & 31, wave = tid >> 5;
  const int wmb = (wave >> 2) * 32, wnb = (wave & 3) * 32;
  v8f acc00 = {}, acc01 = {}, acc10 = {}, acc11 = {};
  for (int t = 0; t < T_STEPS; ++t) {
    const int n = b * T_STEPS + t;
    const float*  an = assigns + (size_t)n * K_DIM * HW;
    const bf16_t* fn = feat    + (size_t)n * D_DIM * HW;
    for (int s0 = 0; s0 < 224; s0 += 32) {
      // A (64 x 32), k = spatial (contiguous); quads fully in/out (196%4==0)
#pragma unroll
      for (int g = tid; g < 512; g += 256) {
        int m = g >> 3, k = (g & 7) << 2;
        int s = s0 + k;
        float4 v = {0.f, 0.f, 0.f, 0.f};
        if (s < HW) v = *(const float4*)(an + (size_t)m * HW + s);
        *(v4bf*)(As + a_base(m, k)) = f2bf4(v.x, v.y, v.z, v.w);
      }
      // B (32 x 128), k contiguous in memory -> vector LDS store too
#pragma unroll
      for (int g = tid; g < 1024; g += 256) {
        int j = g >> 3, k = (g & 7) << 2;
        int s = s0 + k;
        v4bf v = {};
        if (s < HW) v = *(const v4bf*)(fn + (size_t)(d0 + j) * HW + s);
        *(v4bf*)(Bs + b_base(k, j)) = v;
      }
      __syncthreads();
      v16bf a0 = frag16(As, (wmb >> 4),     lane);
      v16bf a1 = frag16(As, (wmb >> 4) + 1, lane);
      v16bf b0 = frag16(Bs, (wnb >> 4),     lane);
      v16bf b1 = frag16(Bs, (wnb >> 4) + 1, lane);
      acc00 = wmma_bf16(a0, b0, acc00);
      acc01 = wmma_bf16(a0, b1, acc01);
      acc10 = wmma_bf16(a1, b0, acc10);
      acc11 = wmma_bf16(a1, b1, acc11);
      __syncthreads();
    }
  }
  const int rm = (lane >> 4) << 3, cn = lane & 15;
#pragma unroll
  for (int mi = 0; mi < 2; ++mi) {
#pragma unroll
    for (int ni = 0; ni < 2; ++ni) {
      v8f a = (mi == 0) ? (ni == 0 ? acc00 : acc01)
                        : (ni == 0 ? acc10 : acc11);
      int mb = wmb + mi * 16 + rm;
      int d  = d0 + wnb + ni * 16 + cn;
#pragma unroll
      for (int r = 0; r < 8; ++r) {
        int k = mb + r;
        float at = 0.0f;
        for (int t = 0; t < T_STEPS; ++t)
          at += asum[(size_t)(b * T_STEPS + t) * K_DIM + k];
        vlad[((size_t)b * K_DIM + k) * D_DIM + d] =
            a[r] - at * centers[(size_t)k * D_DIM + d];
      }
    }
  }
}

// =========================================================================
// Normalization
// =========================================================================
__global__ void k_zero(float* __restrict__ p, int n) {
  int i = blockIdx.x * 256 + threadIdx.x;
  if (i < n) p[i] = 0.0f;
}

__global__ __launch_bounds__(256) void k_norm_rows(const float* __restrict__ vlad,
                                                   float* __restrict__ out,
                                                   float* __restrict__ gacc) {
  const int k = blockIdx.x, b = blockIdx.y;
  const float* row = vlad + ((size_t)b * K_DIM + k) * D_DIM;
  __shared__ float red[256];
  float ss = 0.0f;
  for (int d = threadIdx.x; d < D_DIM; d += 256) { float v = row[d]; ss += v * v; }
  red[threadIdx.x] = ss;
  __syncthreads();
  for (int off = 128; off > 0; off >>= 1) {
    if (threadIdx.x < off) red[threadIdx.x] += red[threadIdx.x + off];
    __syncthreads();
  }
  float nrm = sqrtf(red[0]);
  float inv = 1.0f / fmaxf(nrm, 1e-12f);
  for (int d = threadIdx.x; d < D_DIM; d += 256)
    out[(size_t)b * K_DIM * D_DIM + (size_t)k * D_DIM + d] = row[d] * inv;
  if (threadIdx.x == 0) {
    float c = nrm * inv;
    atomicAdd(gacc + b, c * c);
  }
}

__global__ __launch_bounds__(256) void k_norm_global(float* __restrict__ out,
                                                     const float* __restrict__ gacc) {
  const int b = blockIdx.x;
  float inv = 1.0f / fmaxf(sqrtf(gacc[b]), 1e-12f);
  for (int i = threadIdx.x; i < K_DIM * D_DIM; i += 256)
    out[(size_t)b * K_DIM * D_DIM + i] *= inv;
}

// =========================================================================
// Host-side launcher
// =========================================================================
extern "C" void kernel_launch(void* const* d_in, const int* in_sizes, int n_in,
                              void* d_out, int out_size, void* d_ws, size_t ws_size,
                              hipStream_t stream) {
  (void)in_sizes; (void)n_in; (void)out_size; (void)ws_size;
  const float* x       = (const float*)d_in[0];
  const float* redu_w  = (const float*)d_in[1];
  const float* redu_b  = (const float*)d_in[2];
  const float* share_w = (const float*)d_in[3];
  const float* share_b = (const float*)d_in[4];
  const float* U_z     = (const float*)d_in[5];
  const float* U_r     = (const float*)d_in[6];
  const float* U_h     = (const float*)d_in[7];
  const float* centers = (const float*)d_in[8];
  float* out = (float*)d_out;

  char* ws = (char*)d_ws;
  size_t off = 0;
  auto carve = [&](size_t bytes) -> void* {
    void* p = ws + off;
    off = (off + bytes + 255) & ~(size_t)255;
    return p;
  };
  bf16_t* feat    = (bf16_t*)carve((size_t)NTOT * D_DIM * HW * sizeof(bf16_t));
  float*  wx      = (float*) carve((size_t)T_STEPS * B_IMG * K_DIM * HW * sizeof(float));
  float*  assigns = (float*) carve((size_t)NTOT * K_DIM * HW * sizeof(float));
  float*  zbuf    = (float*) carve((size_t)B_IMG * K_DIM * HW * sizeof(float));
  float*  rhbuf   = (float*) carve((size_t)B_IMG * K_DIM * HW * sizeof(float));
  float*  asum    = (float*) carve((size_t)NTOT * K_DIM * sizeof(float));
  float*  vlad    = (float*) carve((size_t)B_IMG * K_DIM * D_DIM * sizeof(float));
  float*  gacc    = (float*) carve((size_t)B_IMG * sizeof(float));

  k_feat<<<dim3(8, 2, NTOT), 256, 0, stream>>>(x, redu_w, redu_b, feat);
  k_wx<<<dim3(2, NTOT), 256, 0, stream>>>(feat, share_w, share_b, wx);
  const int npix_blocks = (NPIX + 127) / 128;  // 13
  for (int t = 0; t < T_STEPS; ++t) {
    k_gru_zr<<<dim3(npix_blocks), 256, 0, stream>>>(assigns, U_z, U_r, wx, t,
                                                    zbuf, rhbuf);
    k_gru_h<<<dim3(npix_blocks), 256, 0, stream>>>(assigns, U_h, wx, t,
                                                   zbuf, rhbuf);
  }
  k_softmax<<<(NTOT * HW + 255) / 256, 256, 0, stream>>>(assigns);
  k_asum<<<(NTOT * K_DIM + 255) / 256, 256, 0, stream>>>(assigns, asum);
  k_vlad<<<dim3(4, B_IMG), 256, 0, stream>>>(assigns, feat, asum, centers, vlad);
  k_zero<<<1, 256, 0, stream>>>(gacc, B_IMG);
  k_norm_rows<<<dim3(K_DIM, B_IMG), 256, 0, stream>>>(vlad, out, gacc);
  k_norm_global<<<B_IMG, 256, 0, stream>>>(out, gacc);
}
